// LSTMEncoder_41558103556723
// MI455X (gfx1250) — matrix operand order repaired and checked by reading, not compiled, not run on hardware
//
#include <hip/hip_runtime.h>
#include <hip/hip_bf16.h>
#include <stdint.h>

// ---------------------------------------------------------------------------
// 2-layer LSTM for MI455X (gfx1250, wave32).
//   Phase A (per layer): xg = inp @ W_ih + b  via v_wmma_f32_16x16x32_bf16
//   Phase B (per layer): persistent scan, W_hh resident in LDS (128KB/block)
// ---------------------------------------------------------------------------

typedef __bf16 v16bf __attribute__((ext_vector_type(16)));
typedef __bf16 v8bf  __attribute__((ext_vector_type(8)));
typedef float  v8f   __attribute__((ext_vector_type(8)));

#define T_STEPS 8192
#define IN_DIM  512
#define HDIM    1024
#define GDIM    4096   // 4*H

__device__ __forceinline__ unsigned short f2bf(float f) {
  unsigned u = __float_as_uint(f);
  u += 0x7fffu + ((u >> 16) & 1u);     // round-to-nearest-even
  return (unsigned short)(u >> 16);
}

// ---------------- prep kernels ----------------
__global__ void k_f32_to_bf16(const float* __restrict__ src,
                              unsigned short* __restrict__ dst, int n) {
  int i = blockIdx.x * blockDim.x + threadIdx.x;
  if (i < n) dst[i] = f2bf(src[i]);
}

// W: K x GDIM f32 (row major)  ->  Wt: GDIM x K bf16 (row major)
__global__ void k_transpose_bf16(const float* __restrict__ W,
                                 unsigned short* __restrict__ Wt, int K) {
  int i = blockIdx.x * blockDim.x + threadIdx.x;
  if (i < K * GDIM) {
    int n = i / K, k = i - n * K;
    Wt[i] = f2bf(W[(size_t)k * GDIM + n]);
  }
}

__global__ void k_init(unsigned short* h0, unsigned short* h1, unsigned* bar) {
  for (int j = threadIdx.x; j < HDIM; j += blockDim.x) { h0[j] = 0; h1[j] = 0; }
  if (threadIdx.x < 8) bar[threadIdx.x] = 0;
}

// ---------------- WMMA GEMM:  C[M x 4096] = A[M x K] * Bt^T + bias ----------
// A  : M x K bf16 row-major
// Bt : 4096 x K bf16 row-major (pre-transposed weights)
// grid = (GDIM/64, M/128), block = 256 (8 waves; wave = 16 rows x 64 cols)
__global__ __launch_bounds__(256) void k_gemm_bf16(
    const unsigned short* __restrict__ A,
    const unsigned short* __restrict__ Bt,
    const float* __restrict__ bias,
    float* __restrict__ C, int K) {
  const int lane  = threadIdx.x & 31;
  const int wave  = threadIdx.x >> 5;
  const int m0    = blockIdx.y * 128 + wave * 16;
  const int n0    = blockIdx.x * 64;
  const int mlane = lane & 15;
  const int lhalf = lane >> 4;

  v8f acc[4] = {};
  // A fragment addressing (this code's k order, not the ISA table order):
  //   row=mlane, K = 16*lhalf .. +15 (contiguous, matching B)
  const unsigned short* arow = A + (size_t)(m0 + mlane) * K + 16 * lhalf;
  // B fragment: lane owns column n, K = 16*lhalf .. +15 (contiguous in Bt row)
  const unsigned short* brow = Bt + (size_t)(n0 + mlane) * K + 16 * lhalf;

  for (int ko = 0; ko < K; ko += 32) {
    v16bf a = *(const v16bf*)(arow + ko);
#pragma unroll
    for (int nt = 0; nt < 4; ++nt) {
      v16bf b = *(const v16bf*)(brow + (size_t)nt * 16 * K + ko);
      acc[nt] = __builtin_amdgcn_wmma_f32_16x16x32_bf16(
          false, a, false, b, (short)0, acc[nt], false, false);
    }
  }

  // C/D layout: VGPR r -> M = r + 8*lhalf, N = mlane
#pragma unroll
  for (int nt = 0; nt < 4; ++nt) {
    int col = n0 + nt * 16 + mlane;
    float bv = bias[col];
#pragma unroll
    for (int r = 0; r < 8; ++r) {
      int row = m0 + r + 8 * lhalf;
      C[(size_t)row * GDIM + col] = acc[nt][r] + bv;
    }
  }
}

// ---------------- persistent LSTM scan ----------------
struct GBar { unsigned count; unsigned gen; };

__device__ __forceinline__ void grid_barrier(GBar* b, unsigned nblocks) {
  __syncthreads();
  if (threadIdx.x == 0) {
    unsigned g = __hip_atomic_load(&b->gen, __ATOMIC_RELAXED,
                                   __HIP_MEMORY_SCOPE_AGENT);
    unsigned prev = __hip_atomic_fetch_add(&b->count, 1u, __ATOMIC_ACQ_REL,
                                           __HIP_MEMORY_SCOPE_AGENT);
    if (prev == nblocks - 1u) {
      __hip_atomic_store(&b->count, 0u, __ATOMIC_RELAXED,
                         __HIP_MEMORY_SCOPE_AGENT);
      __hip_atomic_fetch_add(&b->gen, 1u, __ATOMIC_RELEASE,
                             __HIP_MEMORY_SCOPE_AGENT);
    } else {
      while (__hip_atomic_load(&b->gen, __ATOMIC_ACQUIRE,
                               __HIP_MEMORY_SCOPE_AGENT) == g) {
        __builtin_amdgcn_s_sleep(2);
      }
    }
  }
  __syncthreads();
}

// WtHH : 4096 x 1024 bf16 (row = gate column)
// xg   : T x 4096 f32 (bias already added)
// Hbf  : (T+1) x 1024 bf16, row 0 zeroed (h_{-1})
// hout : T x 1024 f32 or nullptr
// grid = 64 blocks x 256 threads; block b owns hidden units [16b, 16b+16)
__global__ __launch_bounds__(256) void k_lstm_scan(
    const unsigned short* __restrict__ WtHH,
    const float* __restrict__ xg,
    unsigned short* __restrict__ Hbf,
    float* __restrict__ hout,
    GBar* bar) {
  extern __shared__ char smem[];
  unsigned short* Wsh = (unsigned short*)smem;   // 64 x 1024 bf16 = 128 KB
  unsigned short* hsh = Wsh + 64 * 1024;         // 1024 bf16
  float* gsh = (float*)(hsh + 1024);             // 64 gate pre-activations
  float* csh = gsh + 64;                         // 16 cell states

  const int tid = threadIdx.x;
  const int ub  = blockIdx.x * 16;

  // Stage this block's W_hh slice into LDS (stays resident for all 8192 steps)
  for (int i = tid; i < 64 * 1024 / 8; i += 256) {
    int c   = i >> 7;            // local gate column 0..63
    int off = (i & 127) * 8;     // 8 halfs per uint4
    int gi = c >> 4, u = c & 15;
    size_t gcol = (size_t)(gi * HDIM + ub + u);
    *(uint4*)(Wsh + c * 1024 + off) =
        *(const uint4*)(WtHH + gcol * HDIM + off);
  }
  if (tid < 16) csh[tid] = 0.f;
  __syncthreads();

  const int c = tid >> 2, q = tid & 3;
  const unsigned short* wr = Wsh + c * 1024 + q * 256;
  const unsigned short* hq = hsh + q * 256;

  for (int t = 0; t < T_STEPS; ++t) {
    // broadcast h_{t-1} (row t of Hbf; row 0 is zeros)
    {
      const unsigned short* hrow = Hbf + (size_t)t * HDIM;
      *(uint2*)(hsh + tid * 4) = *(const uint2*)(hrow + tid * 4);
    }

    // Issue this step's xg loads EARLY (independent of h) so they overlap the
    // dot product, and prefetch next step's gate row into cache.
    float xv0 = 0.f, xv1 = 0.f, xv2 = 0.f, xv3 = 0.f;
    if (tid < 16) {
      const float* xgt = xg + (size_t)t * GDIM + ub + tid;
      xv0 = xgt[0];
      xv1 = xgt[HDIM];
      xv2 = xgt[2 * HDIM];
      xv3 = xgt[3 * HDIM];
      if (t + 1 < T_STEPS) {
        const float* xgn = xg + (size_t)(t + 1) * GDIM + ub + tid;
        __builtin_prefetch(xgn, 0, 1);            // global_prefetch_b8
        __builtin_prefetch(xgn + HDIM, 0, 1);
        __builtin_prefetch(xgn + 2 * HDIM, 0, 1);
        __builtin_prefetch(xgn + 3 * HDIM, 0, 1);
      }
    }
    __syncthreads();

    // 64 dot products of length 1024, split 4-ways over K per column
    float s = 0.f;
    auto mac = [&](unsigned w2, unsigned h2) {
      s = fmaf(__uint_as_float(w2 << 16), __uint_as_float(h2 << 16), s);
      s = fmaf(__uint_as_float(w2 & 0xffff0000u),
               __uint_as_float(h2 & 0xffff0000u), s);
    };
#pragma unroll 4
    for (int k = 0; k < 256; k += 8) {
      uint4 w4 = *(const uint4*)(wr + k);
      uint4 h4 = *(const uint4*)(hq + k);
      mac(w4.x, h4.x); mac(w4.y, h4.y); mac(w4.z, h4.z); mac(w4.w, h4.w);
    }
    s += __shfl_xor(s, 1, 32);
    s += __shfl_xor(s, 2, 32);
    if (q == 0) gsh[c] = s;
    __syncthreads();

    if (tid < 16) {
      float zi = gsh[tid]      + xv0;
      float zf = gsh[16 + tid] + xv1;
      float zg = gsh[32 + tid] + xv2;
      float zo = gsh[48 + tid] + xv3;
      float iv = 1.f / (1.f + __expf(-zi));
      float fv = 1.f / (1.f + __expf(-zf));
      float gv = tanhf(zg);
      float ov = 1.f / (1.f + __expf(-zo));
      float cn = fv * csh[tid] + iv * gv;
      csh[tid] = cn;
      float hv = ov * tanhf(cn);
      Hbf[(size_t)(t + 1) * HDIM + ub + tid] = f2bf(hv);
      if (hout) hout[(size_t)t * HDIM + ub + tid] = hv;
      __threadfence();   // make h visible at agent scope before signaling
    }
    grid_barrier(bar, gridDim.x);
  }
}

// ---------------- launcher ----------------
extern "C" void kernel_launch(void* const* d_in, const int* in_sizes, int n_in,
                              void* d_out, int out_size, void* d_ws,
                              size_t ws_size, hipStream_t stream) {
  const float* x    = (const float*)d_in[0];
  const float* Wih0 = (const float*)d_in[1];
  const float* Whh0 = (const float*)d_in[2];
  const float* b0   = (const float*)d_in[3];
  const float* Wih1 = (const float*)d_in[4];
  const float* Whh1 = (const float*)d_in[5];
  const float* b1   = (const float*)d_in[6];
  float* out = (float*)d_out;

  char* ws = (char*)d_ws;
  size_t off = 0;
  auto alloc = [&](size_t bytes) -> char* {
    char* p = ws + off;
    off = (off + bytes + 255) & ~(size_t)255;
    return p;
  };

  GBar* bar             = (GBar*)alloc(256);
  unsigned short* xbf   = (unsigned short*)alloc((size_t)T_STEPS * IN_DIM * 2);
  unsigned short* tWih0 = (unsigned short*)alloc((size_t)GDIM * IN_DIM * 2);
  unsigned short* tWhh0 = (unsigned short*)alloc((size_t)GDIM * HDIM * 2);
  unsigned short* tWih1 = (unsigned short*)alloc((size_t)GDIM * HDIM * 2);
  unsigned short* tWhh1 = (unsigned short*)alloc((size_t)GDIM * HDIM * 2);
  float* xg             = (float*)alloc((size_t)T_STEPS * GDIM * 4);
  unsigned short* H0    = (unsigned short*)alloc((size_t)(T_STEPS + 1) * HDIM * 2);
  unsigned short* H1    = (unsigned short*)alloc((size_t)(T_STEPS + 1) * HDIM * 2);
  (void)in_sizes; (void)n_in; (void)out_size; (void)ws_size;

  k_init<<<1, 256, 0, stream>>>(H0, H1, (unsigned*)bar);

  {
    int n = T_STEPS * IN_DIM;
    k_f32_to_bf16<<<(n + 255) / 256, 256, 0, stream>>>(x, xbf, n);
  }
  k_transpose_bf16<<<(GDIM * IN_DIM + 255) / 256, 256, 0, stream>>>(Wih0, tWih0, IN_DIM);
  k_transpose_bf16<<<(GDIM * HDIM + 255) / 256, 256, 0, stream>>>(Whh0, tWhh0, HDIM);
  k_transpose_bf16<<<(GDIM * HDIM + 255) / 256, 256, 0, stream>>>(Wih1, tWih1, HDIM);
  k_transpose_bf16<<<(GDIM * HDIM + 255) / 256, 256, 0, stream>>>(Whh1, tWhh1, HDIM);

  dim3 ggrid(GDIM / 64, T_STEPS / 128);
  size_t shmem = 64 * 1024 * 2 + 1024 * 2 + (64 + 16) * 4;

  // layer 0
  k_gemm_bf16<<<ggrid, 256, 0, stream>>>(xbf, tWih0, b0, xg, IN_DIM);
  k_lstm_scan<<<64, 256, shmem, stream>>>(tWhh0, xg, H0, nullptr, bar);

  // layer 1 (A = H0 rows 1..T, i.e. skip the zero row)
  k_gemm_bf16<<<ggrid, 256, 0, stream>>>(H0 + HDIM, tWih1, b1, xg, HDIM);
  k_lstm_scan<<<64, 256, shmem, stream>>>(tWhh1, xg, H1, out, bar);
}
